// LinkClassifier_85074712199862
// MI455X (gfx1250) — matrix operand order repaired and checked by reading, not compiled, not run on hardware
//
#include <hip/hip_runtime.h>

typedef __attribute__((ext_vector_type(16))) __bf16 v16bf;
typedef __attribute__((ext_vector_type(8)))  float  v8f;

#define ND      256
#define ED      128
#define HIDN    128
#define NGRAPH  2048
#define KCHUNKS 28      // 896 / 32

// Async global->LDS path (CDNA5): guard with __has_builtin so we always compile.
#if defined(__has_builtin)
#  if __has_builtin(__builtin_amdgcn_global_load_async_to_lds_b128)
#    define HAVE_ASYNC_LDS 1
#  else
#    define HAVE_ASYNC_LDS 0
#  endif
#  if __has_builtin(__builtin_amdgcn_s_wait_asynccnt)
#    define LC_WAIT_ASYNC() __builtin_amdgcn_s_wait_asynccnt(0)
#  else
#    define LC_WAIT_ASYNC() asm volatile("s_wait_asynccnt 0x0" ::: "memory")
#  endif
#else
#  define HAVE_ASYNC_LDS 0
#  define LC_WAIT_ASYNC()
#endif

// Exact operand types for the async builtin (from the compiler diagnostic:
// param 0 is `int __vector(4) __device__ *`, i.e. AS1 pointer to a 16B vector).
typedef int lc_v4i __attribute__((__vector_size__(16)));
typedef __attribute__((address_space(1))) lc_v4i* lc_gp_v4i;
typedef __attribute__((address_space(3))) lc_v4i* lc_lp_v4i;

// ---------------------------------------------------------------------------
// Workspace layout (bytes):
//   [0)                          sums/pooled : NGRAPH*ND f32   (2 MB)
//   [NGRAPH*ND*4)                counts      : NGRAPH f32      (8 KB)
//   [NGRAPH*ND*4 + NGRAPH*4)     w1pk        : KCHUNKS*8192 bf16 (448 KB)
// ---------------------------------------------------------------------------

__global__ void lc_zero_kernel(float* __restrict__ p, int n) {
    int i = blockIdx.x * 256 + threadIdx.x;
    if (i < n) p[i] = 0.0f;
}

// Pack W1 (fp32 row-major [896][128]) into WMMA-B-operand order, bf16 hi + lo.
// Per K-chunk kc: hi block = [nt(8)][lane(32)][j(16)] (4096 elems), then lo.
// Lane half h=lane>>4: j<8 holds K=kc*32+8h+j, j>=8 holds K=kc*32+16+8h+(j-8); col = nt*16+(lane&15).
__global__ void lc_w1_pack_kernel(const float* __restrict__ W1,
                                  __bf16* __restrict__ w1pk) {
    int idx = blockIdx.x * 256 + threadIdx.x;
    if (idx >= KCHUNKS * 8 * 32 * 16) return;
    int j    = idx & 15;
    int lane = (idx >> 4) & 31;
    int nt   = (idx >> 9) & 7;
    int kc   = idx >> 12;
    int K    = kc * 32 + ((lane & 16) >> 1) + (j & 7) + ((j & 8) << 1);
    int col  = nt * 16 + (lane & 15);
    float v  = W1[(size_t)K * HIDN + col];
    __bf16 hb = (__bf16)v;
    __bf16 lb = (__bf16)(v - (float)hb);
    size_t base = (size_t)kc * 8192 + (size_t)nt * 512 + (size_t)lane * 16 + j;
    w1pk[base]        = hb;
    w1pk[base + 4096] = lb;
}

// Segment-sum pooling over sorted batch ids: accumulate runs in registers,
// one atomic per (run, dim).  256 threads = one dim each.
__global__ void lc_pool_sum_kernel(const float* __restrict__ h,
                                   const int* __restrict__ batch,
                                   float* __restrict__ sums,
                                   float* __restrict__ counts, int N) {
    int d  = threadIdx.x;
    int n0 = blockIdx.x * 256;
    if (n0 >= N) return;
    int nend = n0 + 256; if (nend > N) nend = N;
    float acc = 0.0f;
    int cur = batch[n0];
    int runStart = n0;
    for (int n = n0; n < nend; ++n) {
        int b = batch[n];
        if (b != cur) {
            atomicAdd(&sums[(size_t)cur * ND + d], acc);
            if (d == 0) atomicAdd(&counts[cur], (float)(n - runStart));
            acc = 0.0f; cur = b; runStart = n;
        }
        acc += h[(size_t)n * ND + d];
    }
    atomicAdd(&sums[(size_t)cur * ND + d], acc);
    if (d == 0) atomicAdd(&counts[cur], (float)(nend - runStart));
}

__global__ void lc_pool_div_kernel(float* __restrict__ sums,
                                   const float* __restrict__ counts) {
    int i = blockIdx.x * 256 + threadIdx.x;   // NGRAPH*ND total
    int b = i >> 8;                           // ND == 256
    sums[i] = sums[i] / fmaxf(counts[b], 1.0f);
}

// Virtual x row: [ h[s] (256) | h[t] (256) | ea[e] (128) | pooled[batch[s]] (256) ]
__device__ __forceinline__ const float* lc_xsrc(int K, int s, int t, int e, int bs,
                                                const float* __restrict__ h,
                                                const float* __restrict__ ea,
                                                const float* __restrict__ pooled) {
    if (K < ND)           return h      + (size_t)s  * ND + K;
    if (K < 2 * ND)       return h      + (size_t)t  * ND + (K - ND);
    if (K < 2 * ND + ED)  return ea     + (size_t)e  * ED + (K - 2 * ND);
    return pooled + (size_t)bs * ND + (K - 2 * ND - ED);
}

// Stage one 16 KB W1 slice (kc) into LDS buffer `dst`.
__device__ __forceinline__ void lc_stage(const __bf16* __restrict__ w1pk,
                                         __bf16* dst, int kc, int tid) {
    const __bf16* src = w1pk + (size_t)kc * 8192;
#if HAVE_ASYNC_LDS
#pragma unroll
    for (int i = 0; i < 8; ++i) {
        int off = (tid + i * 128) * 8;   // bf16 elems; 16B per lane-chunk
        __builtin_amdgcn_global_load_async_to_lds_b128(
            (lc_gp_v4i)(unsigned long long)(src + off),
            (lc_lp_v4i)(unsigned int)(unsigned long long)(dst + off),
            0, 0);
    }
#else
    const uint4* s4 = (const uint4*)src;
    uint4*       d4 = (uint4*)dst;
#pragma unroll
    for (int i = 0; i < 8; ++i)
        d4[tid + i * 128] = s4[tid + i * 128];
    __builtin_prefetch((const void*)((const char*)src + 16384 + tid * 128), 0, 1);
#endif
}

// Fused edge MLP: 4 waves/block, each wave owns 4 tiles of 16 edges x full H.
// hi/lo bf16 split -> 3x v_wmma_f32_16x16x32_bf16 per (tile, ntile, kchunk).
// Double-buffered async LDS staging of W1 slices.
__global__ __launch_bounds__(128)
void lc_fused_mlp_kernel(const float* __restrict__ h,
                         const int*   __restrict__ ei,
                         const float* __restrict__ ea,
                         const int*   __restrict__ batch,
                         const float* __restrict__ b1,
                         const float* __restrict__ W2,
                         const float* __restrict__ b2,
                         const float* __restrict__ pooled,
                         const __bf16* __restrict__ w1pk,
                         float* __restrict__ out,
                         int E) {
    __shared__ __align__(32) __bf16 ldsW[2][8192];   // 2 x 16 KB double buffer

    const int tid    = threadIdx.x;
    const int wave   = tid >> 5;
    const int lid    = tid & 31;
    const int m      = lid & 15;
    const int hiHalf = (lid >> 4) & 1;

    // Per-tile edge scalars (K-invariant).
    int e_i[4], e_s[4], e_t[4], e_b[4];
#pragma unroll
    for (int t = 0; t < 4; ++t) {
        int e = blockIdx.x * 256 + wave * 64 + t * 16 + m;
        if (e >= E) e = E - 1;                 // clamp; stores guarded later
        e_i[t] = e;
        int s  = ei[e];
        e_s[t] = s;
        e_t[t] = ei[E + e];
        e_b[t] = batch[s];
    }

    v8f acc[4][8];
    const v8f vzero = {0.f, 0.f, 0.f, 0.f, 0.f, 0.f, 0.f, 0.f};
#pragma unroll
    for (int t = 0; t < 4; ++t)
#pragma unroll
        for (int n = 0; n < 8; ++n)
            acc[t][n] = vzero;

    // Prologue: stage chunk 0 into buffer 0.
    lc_stage(w1pk, &ldsW[0][0], 0, tid);

#pragma unroll 1
    for (int kc = 0; kc < KCHUNKS; ++kc) {
        const int buf = kc & 1;
#if HAVE_ASYNC_LDS
        LC_WAIT_ASYNC();          // my slice-portion loads are done
#endif
        __syncthreads();          // whole buffer `buf` is valid block-wide
        if (kc + 1 < KCHUNKS)     // overlap next slice with compute
            lc_stage(w1pk, &ldsW[buf ^ 1][0], kc + 1, tid);

        // ---- A operands for all 4 tiles (ISA 16-bit A 16x32 layout) ----
        const int K0 = kc * 32 + (hiHalf ? 8 : 0);
        v16bf ahi[4], alo[4];
#pragma unroll
        for (int t = 0; t < 4; ++t) {
            const float* p0 = lc_xsrc(K0,      e_s[t], e_t[t], e_i[t], e_b[t], h, ea, pooled);
            const float* p1 = lc_xsrc(K0 + 16, e_s[t], e_t[t], e_i[t], e_b[t], h, ea, pooled);
            float xs[16];
            {
                float4 q;
                q = ((const float4*)p0)[0]; xs[0] = q.x; xs[1] = q.y; xs[2]  = q.z; xs[3]  = q.w;
                q = ((const float4*)p0)[1]; xs[4] = q.x; xs[5] = q.y; xs[6]  = q.z; xs[7]  = q.w;
                q = ((const float4*)p1)[0]; xs[8] = q.x; xs[9] = q.y; xs[10] = q.z; xs[11] = q.w;
                q = ((const float4*)p1)[1]; xs[12]= q.x; xs[13]= q.y; xs[14] = q.z; xs[15] = q.w;
            }
#pragma unroll
            for (int i = 0; i < 16; ++i) {
                float f   = xs[i];
                __bf16 hb = (__bf16)f;
                ahi[t][i] = hb;
                alo[t][i] = (__bf16)(f - (float)hb);
            }
        }

        // ---- B hoisted per N-tile; 4 tiles x 3 split-terms inner ----
#pragma unroll
        for (int n = 0; n < 8; ++n) {
            v16bf bhi = *(const v16bf*)(&ldsW[buf][       n * 512 + lid * 16]);
            v16bf blo = *(const v16bf*)(&ldsW[buf][4096 + n * 512 + lid * 16]);
#pragma unroll
            for (int t = 0; t < 4; ++t) {
                acc[t][n] = __builtin_amdgcn_wmma_f32_16x16x32_bf16(
                    false, ahi[t], false, bhi, (short)0, acc[t][n], false, false);
                acc[t][n] = __builtin_amdgcn_wmma_f32_16x16x32_bf16(
                    false, ahi[t], false, blo, (short)0, acc[t][n], false, false);
                acc[t][n] = __builtin_amdgcn_wmma_f32_16x16x32_bf16(
                    false, alo[t], false, bhi, (short)0, acc[t][n], false, false);
            }
        }
    }

    // Epilogue: bias + ReLU + dot(W2) + cross-lane (N) reduction + store.
    float b1v[8], w2v[8];
#pragma unroll
    for (int n = 0; n < 8; ++n) {
        b1v[n] = b1[n * 16 + m];
        w2v[n] = W2[n * 16 + m];
    }
    const float b2s = b2[0];

#pragma unroll
    for (int t = 0; t < 4; ++t) {
#pragma unroll
        for (int r = 0; r < 8; ++r) {
            float v = 0.0f;
#pragma unroll
            for (int n = 0; n < 8; ++n) {
                float hv = acc[t][n][r] + b1v[n];
                hv = hv > 0.0f ? hv : 0.0f;
                v += hv * w2v[n];
            }
            // C/D layout: lanes 0-15 -> row r, lanes 16-31 -> row r+8; N = lane&15.
            v += __shfl_xor(v, 8, 16);
            v += __shfl_xor(v, 4, 16);
            v += __shfl_xor(v, 2, 16);
            v += __shfl_xor(v, 1, 16);
            if (m == 0) {
                int row = r + hiHalf * 8;
                int e = blockIdx.x * 256 + wave * 64 + t * 16 + row;
                if (e < E) out[e] = v + b2s;
            }
        }
    }
}

extern "C" void kernel_launch(void* const* d_in, const int* in_sizes, int n_in,
                              void* d_out, int out_size, void* d_ws, size_t ws_size,
                              hipStream_t stream) {
    const float* h     = (const float*)d_in[0];
    const int*   ei    = (const int*)  d_in[1];
    const float* ea    = (const float*)d_in[2];
    const int*   batch = (const int*)  d_in[3];
    const float* W1    = (const float*)d_in[4];
    const float* b1    = (const float*)d_in[5];
    const float* W2    = (const float*)d_in[6];
    const float* b2    = (const float*)d_in[7];
    float*       out   = (float*)d_out;

    const int N = in_sizes[0] / ND;     // 100000
    const int E = in_sizes[2] / ED;     // 400000

    char*   ws     = (char*)d_ws;
    float*  sums   = (float*)(ws);                                   // NGRAPH*ND
    float*  counts = (float*)(ws + (size_t)NGRAPH * ND * 4);         // NGRAPH
    __bf16* w1pk   = (__bf16*)(ws + (size_t)NGRAPH * ND * 4 + (size_t)NGRAPH * 4);

    int zn = NGRAPH * ND + NGRAPH;
    lc_zero_kernel<<<(zn + 255) / 256, 256, 0, stream>>>(sums, zn);

    int pk = KCHUNKS * 8 * 32 * 16;
    lc_w1_pack_kernel<<<(pk + 255) / 256, 256, 0, stream>>>(W1, w1pk);

    lc_pool_sum_kernel<<<(N + 255) / 256, 256, 0, stream>>>(h, batch, sums, counts, N);

    lc_pool_div_kernel<<<(NGRAPH * ND) / 256, 256, 0, stream>>>(sums, counts);

    int nblk = (E + 255) / 256;
    lc_fused_mlp_kernel<<<nblk, 128, 0, stream>>>(h, ei, ea, batch, b1, W2, b2,
                                                  sums, w1pk, out, E);
}
